// GWEModel_14259291422731
// MI455X (gfx1250) — matrix operand order repaired and checked by reading, not compiled, not run on hardware
//
#include <hip/hip_runtime.h>
#include <hip/hip_bf16.h>
#include <stdint.h>

typedef __bf16 bf16_t;
typedef __attribute__((ext_vector_type(16))) __bf16 v16bf;
typedef __attribute__((ext_vector_type(8)))  float    v8f;
typedef __attribute__((ext_vector_type(4)))  unsigned int v4u;

#define D_     1024
#define T_     4096      // 2*2048 tokens
#define NL_    8
#define VOCAB_ 32000
#define GH_    256
#define NBLK_  65536     // 16 matrices * 4096 blocks
#define EPS_   1.1920929e-07f

enum { EPI_SILU_BF16 = 0, EPI_SCATTER_W = 1, EPI_RES = 2, EPI_RES_SILU = 3, EPI_OUT = 4 };

template<int N>
__device__ __forceinline__ void async_wait() {
#if __has_builtin(__builtin_amdgcn_s_wait_asynccnt)
    __builtin_amdgcn_s_wait_asynccnt(N);
#else
    asm volatile("s_wait_asynccnt %0" :: "i"(N) : "memory");
#endif
}

// one lane: copy 16 bytes global -> LDS without touching the VGPR data path (ASYNCcnt)
__device__ __forceinline__ void async_copy_b128(const bf16_t* gsrc, unsigned ldsoff) {
    asm volatile("global_load_async_to_lds_b128 %0, %1, off"
                 :: "v"(ldsoff), "v"((unsigned long long)(size_t)gsrc) : "memory");
}

// ---------------------------------------------------------------------------
// WMMA bf16 GEMM:  C[M,N] = A[M,K] * B[N,K]^T   (both row-major along K)
// block tile 128x128, K-step 64, 256 threads = 8 waves in 4x2 of 32x64 tiles.
// Double-buffered LDS, staged with GLOBAL_LOAD_ASYNC_TO_LDS_B128: tile t+1 is
// issued before waiting (ASYNCcnt<=8 drains exactly tile t, in-order retire).
// ---------------------------------------------------------------------------
template<int EPI>
__global__ __launch_bounds__(256, 2) void gemm_bf16_wmma(
    const bf16_t* __restrict__ A, int lda,
    const bf16_t* __restrict__ B, int ldb,
    const float* __restrict__ bias,
    float* __restrict__ Cf, bf16_t* __restrict__ Cb, int ldc,
    int K, float scale)
{
    constexpr int BM = 128, BN = 128, BK = 64, LDT = 80; // 160B row pitch (16B aligned)
    constexpr int ASZ = BM * LDT;                        // halfs per A buffer (20 KB)
    constexpr int BSZ = BN * LDT;                        // halfs per B buffer (20 KB)
    __shared__ bf16_t sA[2 * ASZ];
    __shared__ bf16_t sB[2 * BSZ];

    const int tid  = threadIdx.x;
    const int lane = tid & 31;
    const int wave = tid >> 5;
    const int wm   = wave >> 1;          // 0..3 -> 32 rows each
    const int wn   = wave & 1;           // 0..1 -> 64 cols each
    const int bm   = blockIdx.y * BM;
    const int bn   = blockIdx.x * BN;

    v8f acc[2][4] = {{{}, {}, {}, {}}, {{}, {}, {}, {}}};

    const int sr8 = tid >> 3;            // staging row 0..31
    const int sc8 = tid & 7;             // staging 16B chunk 0..7

    // hoisted addresses: 8 loop-carried global pointers + constant LDS byte offsets
    const bf16_t* gA[4];
    const bf16_t* gB[4];
    unsigned lA[4], lB[4];
#pragma unroll
    for (int i = 0; i < 4; ++i) {
        const int r = sr8 + i * 32;
        gA[i] = A + (size_t)(bm + r) * lda + sc8 * 8;
        gB[i] = B + (size_t)(bn + r) * ldb + sc8 * 8;
        lA[i] = (unsigned)(size_t)(sA + (size_t)r * LDT + sc8 * 8);
        lB[i] = (unsigned)(size_t)(sB + (size_t)r * LDT + sc8 * 8);
    }

    auto issue_tile = [&](int buf) {
        const unsigned ao = (unsigned)buf * (ASZ * 2);   // byte offset of buffer
        const unsigned bo = (unsigned)buf * (BSZ * 2);
#pragma unroll
        for (int i = 0; i < 4; ++i) { async_copy_b128(gA[i], lA[i] + ao); gA[i] += BK; }
#pragma unroll
        for (int i = 0; i < 4; ++i) { async_copy_b128(gB[i], lB[i] + bo); gB[i] += BK; }
    };

    const int nit = K / BK;
    issue_tile(0);                       // prologue: tile 0 in flight

    for (int t = 0; t < nit; ++t) {
        const int buf = t & 1;
        if (t + 1 < nit) {
            issue_tile(buf ^ 1);         // overlap tile t+1 load with tile t compute
            __builtin_prefetch(gA[0], 0, 3);   // warm L2 for tile t+2
            __builtin_prefetch(gB[0], 0, 3);
            async_wait<8>();             // drain exactly tile t (in-order retire)
        } else {
            async_wait<0>();
        }
        __syncthreads();

        const bf16_t* tA = sA + (size_t)buf * ASZ;
        const bf16_t* tB = sB + (size_t)buf * BSZ;

        // A fragment (16x32): lane row = lane&15; chunks at K = kk + {0|8} and +16
        auto ldA = [&](int rbase, int kk) -> v16bf {
            union { v16bf v; v4u u[2]; } f;
            const bf16_t* p = tA + (size_t)(rbase + (lane & 15)) * LDT + kk + ((lane & 16) ? 8 : 0);
            f.u[0] = *(const v4u*)(p);
            f.u[1] = *(const v4u*)(p + 16);
            return f.v;
        };
        // B fragment (32x16): lane col = lane&15; chunks at K = kk + {0|8} and +16
        auto ldB = [&](int rbase, int kk) -> v16bf {
            union { v16bf v; v4u u[2]; } f;
            const bf16_t* p = tB + (size_t)(rbase + (lane & 15)) * LDT + kk + ((lane & 16) ? 8 : 0);
            f.u[0] = *(const v4u*)(p);
            f.u[1] = *(const v4u*)(p + 16);
            return f.v;
        };

#pragma unroll
        for (int kk = 0; kk < BK; kk += 32) {
            v16bf a0 = ldA(wm * 32,      kk);
            v16bf a1 = ldA(wm * 32 + 16, kk);
            v16bf b0 = ldB(wn * 64,      kk);
            v16bf b1 = ldB(wn * 64 + 16, kk);
            v16bf b2 = ldB(wn * 64 + 32, kk);
            v16bf b3 = ldB(wn * 64 + 48, kk);
            acc[0][0] = __builtin_amdgcn_wmma_f32_16x16x32_bf16(false, a0, false, b0, (short)0, acc[0][0], false, false);
            acc[0][1] = __builtin_amdgcn_wmma_f32_16x16x32_bf16(false, a0, false, b1, (short)0, acc[0][1], false, false);
            acc[0][2] = __builtin_amdgcn_wmma_f32_16x16x32_bf16(false, a0, false, b2, (short)0, acc[0][2], false, false);
            acc[0][3] = __builtin_amdgcn_wmma_f32_16x16x32_bf16(false, a0, false, b3, (short)0, acc[0][3], false, false);
            acc[1][0] = __builtin_amdgcn_wmma_f32_16x16x32_bf16(false, a1, false, b0, (short)0, acc[1][0], false, false);
            acc[1][1] = __builtin_amdgcn_wmma_f32_16x16x32_bf16(false, a1, false, b1, (short)0, acc[1][1], false, false);
            acc[1][2] = __builtin_amdgcn_wmma_f32_16x16x32_bf16(false, a1, false, b2, (short)0, acc[1][2], false, false);
            acc[1][3] = __builtin_amdgcn_wmma_f32_16x16x32_bf16(false, a1, false, b3, (short)0, acc[1][3], false, false);
        }
        __syncthreads();
    }

    // epilogue: C/D layout -> row = e + (lane>=16 ? 8 : 0), col = lane&15
    const int hi = (lane & 16) ? 8 : 0;
#pragma unroll
    for (int i = 0; i < 2; ++i)
#pragma unroll
    for (int j = 0; j < 4; ++j) {
        const int row0 = bm + wm * 32 + i * 16 + hi;
        const int col  = bn + wn * 64 + j * 16 + (lane & 15);
        float bb = 0.f;
        if (EPI == EPI_SILU_BF16 || EPI == EPI_SCATTER_W) bb = bias[col];
#pragma unroll
        for (int e = 0; e < 8; ++e) {
            const float c = acc[i][j][e];
            const int row = row0 + e;
            if (EPI == EPI_SILU_BF16) {
                float v = c + bb;
                v = v / (1.f + __expf(-v));
                Cb[(size_t)row * ldc + col] = (bf16_t)v;
            } else if (EPI == EPI_SCATTER_W) {
                // row -> (matrix mi, block bi); col -> (r,c) inside 16x16 block
                float v = c + bb;
                const int mi = row >> 12, bi = row & 4095;
                const int rb = bi >> 6, cb2 = bi & 63;
                const size_t w = ((size_t)mi << 20)
                               + (size_t)((rb << 4) + (col >> 4)) * D_
                               + (size_t)((cb2 << 4) + (col & 15));
                Cb[w] = (bf16_t)v;
            } else if (EPI == EPI_RES) {
                const size_t idx = (size_t)row * ldc + col;
                Cf[idx] += scale * c;
            } else if (EPI == EPI_RES_SILU) {
                const size_t idx = (size_t)row * ldc + col;
                const float v = c / (1.f + __expf(-c));
                Cf[idx] += scale * v;
            } else { // EPI_OUT
                Cf[(size_t)row * ldc + col] = c;
            }
        }
    }
}

// ---------------------------------------------------------------------------
// RMSNorm: one block per row, 256 threads x 4 elems, emit bf16
// ---------------------------------------------------------------------------
__global__ __launch_bounds__(256) void rmsnorm_kernel(
    const float* __restrict__ x, const float* __restrict__ w, bf16_t* __restrict__ out)
{
    const int row = blockIdx.x;
    const int tid = threadIdx.x;
    const float4 v = *(const float4*)(x + (size_t)row * D_ + tid * 4);
    float ss = v.x * v.x + v.y * v.y + v.z * v.z + v.w * v.w;
#pragma unroll
    for (int off = 16; off; off >>= 1) ss += __shfl_xor(ss, off, 32);
    __shared__ float red[8];
    if ((tid & 31) == 0) red[tid >> 5] = ss;
    __syncthreads();
    float tot = 0.f;
#pragma unroll
    for (int i = 0; i < 8; ++i) tot += red[i];
    const float inv = rsqrtf(tot * (1.0f / D_) + EPS_);
    const float4 wv = *(const float4*)(w + tid * 4);
    union { bf16_t h[4]; uint64_t u; } o;
    o.h[0] = (bf16_t)(v.x * inv * wv.x);
    o.h[1] = (bf16_t)(v.y * inv * wv.y);
    o.h[2] = (bf16_t)(v.z * inv * wv.z);
    o.h[3] = (bf16_t)(v.w * inv * wv.w);
    *(uint64_t*)(out + (size_t)row * D_ + tid * 4) = o.u;
}

// embedding gather: one block per token
__global__ __launch_bounds__(256) void gather_kernel(
    const int* __restrict__ tokens, const float* __restrict__ embed, float* __restrict__ x)
{
    const int t = blockIdx.x;
    const int tok = tokens[t];
    const float4 v = *(const float4*)(embed + (size_t)tok * D_ + threadIdx.x * 4);
    *(float4*)(x + (size_t)t * D_ + threadIdx.x * 4) = v;
}

// fourier encoding rows: [NBLK, 64] bf16 (52 real + 12 zero pad)
__global__ __launch_bounds__(256) void enc_kernel(bf16_t* __restrict__ ENC)
{
    const int i = blockIdx.x * 256 + threadIdx.x;
    if (i >= NBLK_) return;
    const int mi = i >> 12, bi = i & 4095;
    const int li = mi >> 1;
    const float ft = (mi & 1) ? 0.5f : 0.0f;
    const float c4[4] = { li * (1.0f / 7.0f), (float)(bi >> 6) * (1.0f / 63.0f),
                          (float)(bi & 63) * (1.0f / 63.0f), ft };
    bf16_t row[64];
    int p = 0;
#pragma unroll
    for (int j = 0; j < 4; ++j) row[p++] = (bf16_t)c4[j];
#pragma unroll
    for (int f = 0; f < 6; ++f) {
        const float m = (float)(1 << f) * 3.14159265358979323846f;
#pragma unroll
        for (int j = 0; j < 4; ++j) row[p++] = (bf16_t)__sinf(m * c4[j]);
#pragma unroll
        for (int j = 0; j < 4; ++j) row[p++] = (bf16_t)__cosf(m * c4[j]);
    }
    while (p < 64) row[p++] = (bf16_t)0.0f;
    bf16_t* dst = ENC + (size_t)i * 64;
#pragma unroll
    for (int q = 0; q < 8; ++q) *(v4u*)(dst + q * 8) = *(const v4u*)(row + q * 8);
}

// f32 -> bf16 convert
__global__ __launch_bounds__(256) void cvt_f32_bf16(
    const float* __restrict__ s, bf16_t* __restrict__ d, size_t n)
{
    const size_t i = (size_t)blockIdx.x * 256 + threadIdx.x;
    if (i < n) d[i] = (bf16_t)s[i];
}

// pad g_w0 [256,52] -> [256,64] bf16
__global__ __launch_bounds__(64) void cvt_w0_pad(const float* __restrict__ s, bf16_t* __restrict__ d)
{
    const int r = blockIdx.x, c = threadIdx.x;
    d[(size_t)r * 64 + c] = (c < 52) ? (bf16_t)s[(size_t)r * 52 + c] : (bf16_t)0.0f;
}

// ---------------------------------------------------------------------------
extern "C" void kernel_launch(void* const* d_in, const int* in_sizes, int n_in,
                              void* d_out, int out_size, void* d_ws, size_t ws_size,
                              hipStream_t stream) {
    (void)in_sizes; (void)n_in; (void)out_size; (void)ws_size;
    const int*   tokens   = (const int*)  d_in[0];
    const float* embed    = (const float*)d_in[1];
    const float* lm_head  = (const float*)d_in[2];
    const float* final_nw = (const float*)d_in[3];
    const float* norm1_w  = (const float*)d_in[4];
    const float* norm2_w  = (const float*)d_in[5];
    const float* g_w0 = (const float*)d_in[6];   const float* g_b0 = (const float*)d_in[7];
    const float* g_w1 = (const float*)d_in[8];   const float* g_b1 = (const float*)d_in[9];
    const float* g_w2 = (const float*)d_in[10];  const float* g_b2 = (const float*)d_in[11];
    const float* g_w3 = (const float*)d_in[12];  const float* g_b3 = (const float*)d_in[13];
    float* out = (float*)d_out;

    // workspace layout (~200 MB)
    char* w = (char*)d_ws;
    bf16_t* Wg   = (bf16_t*)w; w += (size_t)16 * 1024 * 1024 * 2;   // 16 generated matrices, bf16
    float*  x    = (float*) w; w += (size_t)T_ * D_ * 4;            // fp32 residual
    bf16_t* hb   = (bf16_t*)w; w += (size_t)T_ * D_ * 2;            // normed activations bf16
    bf16_t* ENC  = (bf16_t*)w; w += (size_t)NBLK_ * 64 * 2;
    bf16_t* H1   = (bf16_t*)w; w += (size_t)NBLK_ * GH_ * 2;
    bf16_t* H2   = (bf16_t*)w; w += (size_t)NBLK_ * GH_ * 2;
    bf16_t* w0p  = (bf16_t*)w; w += (size_t)GH_ * 64 * 2;
    bf16_t* w1b  = (bf16_t*)w; w += (size_t)GH_ * GH_ * 2;
    bf16_t* w2b  = (bf16_t*)w; w += (size_t)GH_ * GH_ * 2;
    bf16_t* w3b  = (bf16_t*)w; w += (size_t)GH_ * GH_ * 2;
    bf16_t* lmhb = (bf16_t*)w; w += (size_t)VOCAB_ * D_ * 2;

    // --- convert weights to bf16 ---
    cvt_w0_pad<<<GH_, 64, 0, stream>>>(g_w0, w0p);
    cvt_f32_bf16<<<(GH_ * GH_ + 255) / 256, 256, 0, stream>>>(g_w1, w1b, (size_t)GH_ * GH_);
    cvt_f32_bf16<<<(GH_ * GH_ + 255) / 256, 256, 0, stream>>>(g_w2, w2b, (size_t)GH_ * GH_);
    cvt_f32_bf16<<<(GH_ * GH_ + 255) / 256, 256, 0, stream>>>(g_w3, w3b, (size_t)GH_ * GH_);
    {
        const size_t n = (size_t)VOCAB_ * D_;
        cvt_f32_bf16<<<(unsigned)((n + 255) / 256), 256, 0, stream>>>(lm_head, lmhb, n);
    }

    // --- generate the 16 weight matrices via batched MLP (all WMMA) ---
    enc_kernel<<<NBLK_ / 256, 256, 0, stream>>>(ENC);
    {
        dim3 g(GH_ / 128, NBLK_ / 128);  // N=256, M=65536
        gemm_bf16_wmma<EPI_SILU_BF16><<<g, 256, 0, stream>>>(ENC, 64,  w0p, 64,  g_b0, nullptr, H1, GH_, 64,  0.f);
        gemm_bf16_wmma<EPI_SILU_BF16><<<g, 256, 0, stream>>>(H1, GH_,  w1b, GH_, g_b1, nullptr, H2, GH_, GH_, 0.f);
        gemm_bf16_wmma<EPI_SILU_BF16><<<g, 256, 0, stream>>>(H2, GH_,  w2b, GH_, g_b2, nullptr, H1, GH_, GH_, 0.f);
        gemm_bf16_wmma<EPI_SCATTER_W><<<g, 256, 0, stream>>>(H1, GH_,  w3b, GH_, g_b3, nullptr, Wg, 0,   GH_, 0.f);
    }

    // --- transformer body ---
    gather_kernel<<<T_, 256, 0, stream>>>(tokens, embed, x);
    const dim3 gl(D_ / 128, T_ / 128);   // N=1024, M=4096
    for (int li = 0; li < NL_; ++li) {
        rmsnorm_kernel<<<T_, 256, 0, stream>>>(x, norm1_w + (size_t)li * D_, hb);
        gemm_bf16_wmma<EPI_RES><<<gl, 256, 0, stream>>>(
            hb, D_, Wg + (size_t)(li * 2) * D_ * D_, D_, nullptr, x, nullptr, D_, D_, 0.1f);
        rmsnorm_kernel<<<T_, 256, 0, stream>>>(x, norm2_w + (size_t)li * D_, hb);
        gemm_bf16_wmma<EPI_RES_SILU><<<gl, 256, 0, stream>>>(
            hb, D_, Wg + (size_t)(li * 2 + 1) * D_ * D_, D_, nullptr, x, nullptr, D_, D_, 0.1f);
    }

    // --- final norm + LM head ---
    rmsnorm_kernel<<<T_, 256, 0, stream>>>(x, final_nw, hb);
    {
        dim3 g(VOCAB_ / 128, T_ / 128);  // N=32000, M=4096
        gemm_bf16_wmma<EPI_OUT><<<g, 256, 0, stream>>>(
            hb, D_, lmhb, D_, nullptr, out, nullptr, VOCAB_, D_, 1.0f);
    }
}
